// DCE_TMT_65481071395030
// MI455X (gfx1250) — compile-verified, matrix operand order checked by reading, not run on hardware
//
#include <hip/hip_runtime.h>
#include <hip/hip_bf16.h>

// ---------------------------------------------------------------------------
// Problem constants (from reference): B=16, L=256, D=768, NH=12, DK=64, NL=2
// ---------------------------------------------------------------------------
#define B_   16
#define L_   256
#define D_   768
#define NH_  12
#define DK_  64
#define NL_  2
#define M_   (B_*L_)      // 4096 rows
#define FF_  (2*D_)       // 1536
#define NEGV (-1e9f)

typedef __attribute__((ext_vector_type(16))) _Float16 v16h;
typedef __attribute__((ext_vector_type(8)))  _Float16 h8;
typedef __attribute__((ext_vector_type(8)))  float    v8f;

// 128-bit pointee type required by the async-LDS builtins (signature vV4i*1V4i*3IiIi)
typedef int i4_t __attribute__((ext_vector_type(4)));
typedef __attribute__((address_space(1))) i4_t as1_i4;
typedef __attribute__((address_space(3))) i4_t as3_i4;

// ---- CDNA5 async global->LDS copy (ASYNCcnt path) --------------------------
__device__ __forceinline__ void async_copy_b128(const _Float16* g, _Float16* l) {
#if __has_builtin(__builtin_amdgcn_global_load_async_to_lds_b128)
    __builtin_amdgcn_global_load_async_to_lds_b128(
        (as1_i4*)(unsigned long long)(g),
        (as3_i4*)(unsigned long long)(l), 0, 0);
#elif defined(__gfx1250__)
    unsigned lds_off = (unsigned)(unsigned long long)(l);
    unsigned long long ga = (unsigned long long)g;
    asm volatile("global_load_async_to_lds_b128 %0, %1, off"
                 :: "v"(lds_off), "v"(ga) : "memory");
#else
    *(h8*)l = *(const h8*)g;
#endif
}

__device__ __forceinline__ void wait_async_all() {
#if __has_builtin(__builtin_amdgcn_s_wait_asynccnt)
    __builtin_amdgcn_s_wait_asynccnt(0);
#elif defined(__gfx1250__)
    asm volatile("s_wait_asynccnt 0" ::: "memory");
#endif
}

// ===========================================================================
// WMMA GEMM: C[M,N] = act( alpha * A[M,K](f16) * Bt[N,K](f16)^T + bias + res )
// B is always supplied TRANSPOSED (Bt is N x K row-major) so both A and B
// tiles stage as contiguous 64x32 half tiles via GLOBAL_LOAD_ASYNC_TO_LDS_B128
// and both fragment gathers are contiguous ds_load_b128 pairs.
// Block = 128 threads (4 waves). Block tile 64x64, wave tile 32x32 (2x2 WMMA),
// K-step 32, double-buffered LDS with one s_wait_asynccnt per tile.
// Batched via blockIdx.z with 2-level stride decomposition (z -> z/nH, z%nH).
// Requires: M%64==0, N%64==0, K%32==0 (true for all uses here).
// ===========================================================================
template<bool HAS_BIAS, bool HAS_RES, int ACT>
__global__ __launch_bounds__(128)
void gemm_wmma(float* __restrict__ Cp,
               const _Float16* __restrict__ Ap,
               const _Float16* __restrict__ Btp,
               const float* __restrict__ bias,
               const float* __restrict__ resp,
               int M, int N, int K, int lda, int ldb, int ldc,
               long long sA0, long long sA1,
               long long sB0, long long sB1,
               long long sC0, long long sC1,
               long long sR0, long long sR1,
               int nH, float alpha)
{
    __shared__ _Float16 As[2][64][40] __attribute__((aligned(16)));
    __shared__ _Float16 Bs[2][64][40] __attribute__((aligned(16)));

    const int t    = threadIdx.x;
    const int lane = t & 31;
    const int wid  = t >> 5;
    const int wm   = (wid >> 1) * 32;   // wave row offset in block tile
    const int wn   = (wid & 1) * 32;    // wave col offset in block tile

    const int z  = blockIdx.z;
    const int zb = z / nH, zh = z % nH;

    const _Float16* A = Ap + (long long)zb * sA0 + (long long)zh * sA1
                           + (long long)(blockIdx.y * 64) * lda;
    const _Float16* Bt = Btp + (long long)zb * sB0 + (long long)zh * sB1
                             + (long long)(blockIdx.x * 64) * ldb;
    const int col0  = blockIdx.x * 64;
    const int row0g = blockIdx.y * 64;

    const int hlf = lane >> 4;   // which 16-lane half
    const int mr  = lane & 15;   // row/col within 16

    v8f acc[2][2] = {};

    // ---- cooperative tile staging: 2 x (64 rows x 32 halfs), async b128 ---
    auto stage = [&](int buf, int k0) {
        #pragma unroll
        for (int c = t; c < 256; c += 128) {
            int r  = c >> 2;
            int kc = (c & 3) * 8;
            async_copy_b128(&A[(long long)r * lda + k0 + kc], &As[buf][r][kc]);
        }
        #pragma unroll
        for (int c = t; c < 256; c += 128) {
            int r  = c >> 2;
            int kc = (c & 3) * 8;
            async_copy_b128(&Bt[(long long)r * ldb + k0 + kc], &Bs[buf][r][kc]);
        }
    };

    const int nk = K >> 5;
    stage(0, 0);

    for (int kt = 0; kt < nk; ++kt) {
        const int buf = kt & 1;
        wait_async_all();        // my async loads for this tile done
        __syncthreads();         // everyone's loads done; prev compute done
        if (kt + 1 < nk) stage(buf ^ 1, (kt + 1) << 5);

        // ---- gather fragments per ISA 16-bit A/B lane layout --------------
        // Lane holds 8-half contiguous K-chunks: [8*hlf .. +8) and [16+8*hlf .. +8)
        v16h af[2], bf[2];
        #pragma unroll
        for (int sm = 0; sm < 2; ++sm) {
            const int rb = wm + sm * 16 + mr;
            #pragma unroll
            for (int v = 0; v < 8; ++v) {
                int kb = ((v & 4) << 2) + 8 * hlf + ((v & 3) << 1);
                af[sm][2 * v]     = As[buf][rb][kb];
                af[sm][2 * v + 1] = As[buf][rb][kb + 1];
            }
        }
        #pragma unroll
        for (int sn = 0; sn < 2; ++sn) {
            const int cb = wn + sn * 16 + mr;
            #pragma unroll
            for (int v = 0; v < 8; ++v) {
                int kb = ((v & 4) << 2) + 8 * hlf + ((v & 3) << 1);
                bf[sn][2 * v]     = Bs[buf][cb][kb];
                bf[sn][2 * v + 1] = Bs[buf][cb][kb + 1];
            }
        }
        #pragma unroll
        for (int sm = 0; sm < 2; ++sm)
            #pragma unroll
            for (int sn = 0; sn < 2; ++sn)
                acc[sm][sn] = __builtin_amdgcn_wmma_f32_16x16x32_f16(
                    false, af[sm], false, bf[sn],
                    (short)0, acc[sm][sn], false, false);
    }

    // ---- epilogue ---------------------------------------------------------
    float* C = Cp + (long long)zb * sC0 + (long long)zh * sC1;
    const float* R = HAS_RES ? (resp + (long long)zb * sR0 + (long long)zh * sR1)
                             : nullptr;
    #pragma unroll
    for (int sm = 0; sm < 2; ++sm) {
        #pragma unroll
        for (int sn = 0; sn < 2; ++sn) {
            #pragma unroll
            for (int r = 0; r < 8; ++r) {
                int row = row0g + wm + sm * 16 + hlf * 8 + r;
                int col = col0 + wn + sn * 16 + mr;
                float v = acc[sm][sn][r] * alpha;
                if (HAS_BIAS) v += bias[col];
                if (HAS_RES)  v += R[(long long)row * ldc + col];
                if (ACT == 1) v = fmaxf(v, 0.0f);
                C[(long long)row * ldc + col] = v;
            }
        }
    }
}

// ===========================================================================
// Elementwise / reduction helpers
// ===========================================================================
__global__ void cvt_kernel(_Float16* __restrict__ d, const float* __restrict__ s,
                           long long n) {
    for (long long i = (long long)blockIdx.x * blockDim.x + threadIdx.x;
         i < n; i += (long long)gridDim.x * blockDim.x)
        d[i] = (_Float16)s[i];
}

// Transpose-convert: src[b][r][c] (f32) -> dst[b][c][r] (f16)
__global__ void cvt_t3_kernel(_Float16* __restrict__ d, const float* __restrict__ s,
                              int nb, int R, int C) {
    long long n = (long long)nb * R * C;
    for (long long i = (long long)blockIdx.x * blockDim.x + threadIdx.x;
         i < n; i += (long long)gridDim.x * blockDim.x) {
        long long b = i / ((long long)R * C);
        long long rem = i % ((long long)R * C);
        int r = (int)(rem / C), c = (int)(rem % C);
        d[b * (long long)R * C + (long long)c * R + r] = (_Float16)s[i];
    }
}

// Transpose-scale-convert: dst[b][c][r] = (f16)(rowscale[b*R+r] * src[b][r][c])
__global__ void scale_cvt_t3_kernel(_Float16* __restrict__ d,
                                    const float* __restrict__ s,
                                    const float* __restrict__ rowscale,
                                    int nb, int R, int C) {
    long long n = (long long)nb * R * C;
    for (long long i = (long long)blockIdx.x * blockDim.x + threadIdx.x;
         i < n; i += (long long)gridDim.x * blockDim.x) {
        long long b = i / ((long long)R * C);
        long long rem = i % ((long long)R * C);
        int r = (int)(rem / C), c = (int)(rem % C);
        float v = rowscale[b * R + r] * s[i];
        d[b * (long long)R * C + (long long)c * R + r] = (_Float16)v;
    }
}

__global__ void copy_kernel(float* __restrict__ d, const float* __restrict__ s,
                            long long n) {
    for (long long i = (long long)blockIdx.x * blockDim.x + threadIdx.x;
         i < n; i += (long long)gridDim.x * blockDim.x)
        d[i] = s[i];
}

template<bool RELU>
__global__ void scale_cvt_kernel(_Float16* __restrict__ d,
                                 const float* __restrict__ s,
                                 const float* __restrict__ rowscale,
                                 int len, long long n) {
    for (long long i = (long long)blockIdx.x * blockDim.x + threadIdx.x;
         i < n; i += (long long)gridDim.x * blockDim.x) {
        long long row = i / len;
        float v = rowscale[row] * s[i];
        if (RELU) v = fmaxf(v, 0.0f);
        d[i] = (_Float16)v;
    }
}

// aspect_outs[b,d] = sum_l h[b,l,d]*am[b,l] / sum_l am[b,l]
__global__ void aspect_mean_kernel(float* __restrict__ out,
                                   const float* __restrict__ h,
                                   const int* __restrict__ am) {
    int idx = blockIdx.x * blockDim.x + threadIdx.x;
    if (idx >= B_ * D_) return;
    int b = idx / D_, d = idx % D_;
    float num = 0.f, den = 0.f;
    for (int l = 0; l < L_; ++l) {
        float m = (float)am[b * L_ + l];
        num += h[((long long)(b * L_ + l)) * D_ + d] * m;
        den += m;
    }
    out[idx] = num / den;
}

// asp[b,j] = aspect[b,:] @ dense_w[:,j] + dense_b[j]
__global__ void asp_kernel(float* __restrict__ out,
                           const float* __restrict__ aspect,
                           const float* __restrict__ dw,
                           const float* __restrict__ db) {
    int idx = blockIdx.x * blockDim.x + threadIdx.x;
    if (idx >= B_ * DK_) return;
    int b = idx / DK_, j = idx % DK_;
    float s = 0.f;
    for (int k = 0; k < D_; ++k) s += aspect[b * D_ + k] * dw[k * DK_ + j];
    out[idx] = s + db[j];
}

// aw[b,h,j] = sum_k asp[b,k] * weight_m[h,k,j]
__global__ void aw_kernel(float* __restrict__ out,
                          const float* __restrict__ asp,
                          const float* __restrict__ wm) {
    int idx = blockIdx.x * blockDim.x + threadIdx.x;
    if (idx >= B_ * NH_ * DK_) return;
    int b = idx / (NH_ * DK_);
    int h = (idx / DK_) % NH_;
    int j = idx % DK_;
    float s = 0.f;
    for (int k = 0; k < DK_; ++k)
        s += asp[b * DK_ + k] * wm[(h * DK_ + k) * DK_ + j];
    out[idx] = s;
}

// asc[b,h,l] = tanh( sum_j aw[b,h,j] * k[b,l,h*64+j] + bias_m )
__global__ void asc_kernel(float* __restrict__ out,
                           const float* __restrict__ aw,
                           const float* __restrict__ kf,
                           const float* __restrict__ bias_m) {
    int idx = blockIdx.x * blockDim.x + threadIdx.x;
    if (idx >= B_ * NH_ * L_) return;
    int b = idx / (NH_ * L_);
    int h = (idx / L_) % NH_;
    int l = idx % L_;
    float s = 0.f;
    const float* kr = kf + ((long long)(b * L_ + l)) * D_ + h * DK_;
    const float* ar = aw + (b * NH_ + h) * DK_;
    for (int j = 0; j < DK_; ++j) s += ar[j] * kr[j];
    out[idx] = tanhf(s + bias_m[0]);
}

// Softmax over rows of scores[B,NH,L,L] in place.
__global__ void softmax_kernel(float* __restrict__ s,
                               const float* __restrict__ asc, // nullable
                               const unsigned char* __restrict__ mask,
                               int use_short) {
    __shared__ float sh[L_];
    const int t   = threadIdx.x;
    const int row = blockIdx.x;              // (b*NH+h)*L + i
    const int b   = row / (NH_ * L_);
    const int bh  = row / L_;
    const int i   = row % L_;
    float* p = s + (long long)row * L_;
    float v = p[t];
    if (asc) v += asc[bh * L_ + t];
    if (use_short) v -= fabsf((float)(i - t));
    if (!mask[b * L_ + t]) v = NEGV;

    sh[t] = v; __syncthreads();
    for (int st = L_ / 2; st > 0; st >>= 1) {
        if (t < st) sh[t] = fmaxf(sh[t], sh[t + st]);
        __syncthreads();
    }
    float mx = sh[0]; __syncthreads();
    float e = __expf(v - mx);
    sh[t] = e; __syncthreads();
    for (int st = L_ / 2; st > 0; st >>= 1) {
        if (t < st) sh[t] += sh[t + st];
        __syncthreads();
    }
    p[t] = e / sh[0];
}

// wamean[b,i,j] = mean_h weight_adj[b,h,i,j]
__global__ void wamean_kernel(float* __restrict__ out,
                              const float* __restrict__ wa) {
    long long idx = (long long)blockIdx.x * blockDim.x + threadIdx.x;
    if (idx >= (long long)B_ * L_ * L_) return;
    long long b = idx / (L_ * L_);
    long long rem = idx % (L_ * L_);
    float s = 0.f;
    for (int h = 0; h < NH_; ++h)
        s += wa[(b * NH_ + h) * (long long)(L_ * L_) + rem];
    out[idx] = s * (1.0f / NH_);
}

// Build Hm[B,L,2L] (f32), Hm16 and HmT16 ([B,2L,L]) in one pass.
__global__ void build_hm_kernel(float* __restrict__ Hm,
                                _Float16* __restrict__ Hm16,
                                _Float16* __restrict__ HmT16,
                                const float* __restrict__ wamean,
                                const int* __restrict__ adj) {
    long long idx = (long long)blockIdx.x * blockDim.x + threadIdx.x;
    if (idx >= (long long)B_ * L_ * (2 * L_)) return;
    int e = (int)(idx % (2 * L_));
    int l = (int)((idx / (2 * L_)) % L_);
    int b = (int)(idx / ((long long)L_ * 2 * L_));
    float v;
    if (e < L_) v = wamean[((long long)b * L_ + l) * L_ + e];
    else        v = (float)adj[((long long)b * L_ + l) * L_ + (e - L_)];
    Hm[idx] = v;
    Hm16[idx] = (_Float16)v;
    HmT16[((long long)b * (2 * L_) + e) * L_ + l] = (_Float16)v;
}

__global__ void dv_kernel(float* __restrict__ Dv, const float* __restrict__ Hm) {
    int idx = blockIdx.x * blockDim.x + threadIdx.x;
    if (idx >= B_ * L_) return;
    const float* r = Hm + (long long)idx * (2 * L_);
    float s = 0.f;
    for (int e = 0; e < 2 * L_; ++e) s += r[e];
    Dv[idx] = 1.0f / (s + 1e-9f);
}

__global__ void de_kernel(float* __restrict__ De, const float* __restrict__ Hm) {
    int idx = blockIdx.x * blockDim.x + threadIdx.x;
    if (idx >= B_ * 2 * L_) return;
    int b = idx / (2 * L_), e = idx % (2 * L_);
    float s = 0.f;
    for (int l = 0; l < L_; ++l)
        s += Hm[((long long)b * L_ + l) * (2 * L_) + e];
    De[idx] = 1.0f / (s + 1e-9f);
}

// LayerNorm over last dim (=768). y = ln(x [+res]) * g + b [relu] [+ post]
template<bool RELU>
__global__ void ln_kernel(float* __restrict__ out,
                          const float* __restrict__ in,
                          const float* __restrict__ res,   // nullable, pre-add
                          const float* __restrict__ g,
                          const float* __restrict__ b,
                          const float* __restrict__ post)  // nullable, post-add
{
    __shared__ float sh[256];
    const long long row = blockIdx.x;
    const int t = threadIdx.x;
    const float* x = in + row * D_;
    float v0 = x[t], v1 = x[t + 256], v2 = x[t + 512];
    if (res) {
        const float* r = res + row * D_;
        v0 += r[t]; v1 += r[t + 256]; v2 += r[t + 512];
    }
    sh[t] = v0 + v1 + v2; __syncthreads();
    for (int st = 128; st > 0; st >>= 1) {
        if (t < st) sh[t] += sh[t + st];
        __syncthreads();
    }
    float m = sh[0] * (1.0f / D_); __syncthreads();
    float d0 = v0 - m, d1 = v1 - m, d2 = v2 - m;
    sh[t] = d0 * d0 + d1 * d1 + d2 * d2; __syncthreads();
    for (int st = 128; st > 0; st >>= 1) {
        if (t < st) sh[t] += sh[t + st];
        __syncthreads();
    }
    float inv = rsqrtf(sh[0] * (1.0f / D_) + 1e-5f);
    float y0 = d0 * inv * g[t]       + b[t];
    float y1 = d1 * inv * g[t + 256] + b[t + 256];
    float y2 = d2 * inv * g[t + 512] + b[t + 512];
    if (RELU) { y0 = fmaxf(y0, 0.f); y1 = fmaxf(y1, 0.f); y2 = fmaxf(y2, 0.f); }
    if (post) {
        const float* p = post + row * D_;
        y0 += p[t]; y1 += p[t + 256]; y2 += p[t + 512];
    }
    float* o = out + row * D_;
    o[t] = y0; o[t + 256] = y1; o[t + 512] = y2;
}

// ===========================================================================
// Host orchestration
// ===========================================================================
static void cvt(hipStream_t st, _Float16* d, const float* s, long long n) {
    int blocks = (int)(((n + 255) / 256 > 2048) ? 2048 : (n + 255) / 256);
    cvt_kernel<<<blocks, 256, 0, st>>>(d, s, n);
}
static void cvt_t3(hipStream_t st, _Float16* d, const float* s, int nb, int R, int C) {
    long long n = (long long)nb * R * C;
    int blocks = (int)(((n + 255) / 256 > 2048) ? 2048 : (n + 255) / 256);
    cvt_t3_kernel<<<blocks, 256, 0, st>>>(d, s, nb, R, C);
}

template<bool HAS_BIAS, bool HAS_RES, int ACT>
static void run_gemm(hipStream_t st, float* C, const _Float16* A,
                     const _Float16* Bt, const float* bias, const float* res,
                     int M, int N, int K, int lda, int ldb, int ldc,
                     long long sA0, long long sA1, long long sB0, long long sB1,
                     long long sC0, long long sC1, long long sR0, long long sR1,
                     int nH, int batches, float alpha) {
    dim3 grid(N / 64, M / 64, batches), blk(128);
    gemm_wmma<HAS_BIAS, HAS_RES, ACT><<<grid, blk, 0, st>>>(
        C, A, Bt, bias, res, M, N, K, lda, ldb, ldc,
        sA0, sA1, sB0, sB1, sC0, sC1, sR0, sR1, nH, alpha);
}

extern "C" void kernel_launch(void* const* d_in, const int* in_sizes, int n_in,
                              void* d_out, int out_size, void* d_ws, size_t ws_size,
                              hipStream_t stream) {
    // ---- inputs (setup_inputs order) -------------------------------------
    const float* x       = (const float*)d_in[0];
    const int*   adj     = (const int*)d_in[1];
    const unsigned char* mask = (const unsigned char*)d_in[2]; // jnp bool = 1B
    const int*   amask   = (const int*)d_in[3];
    const float* W_in    = (const float*)d_in[4];
    const float* b_in    = (const float*)d_in[5];
    const float* q_w     = (const float*)d_in[6];
    const float* q_b     = (const float*)d_in[7];
    const float* k_w     = (const float*)d_in[8];
    const float* k_b     = (const float*)d_in[9];
    const float* dense_w = (const float*)d_in[10];
    const float* dense_b = (const float*)d_in[11];
    const float* weight_m= (const float*)d_in[12];
    const float* bias_m  = (const float*)d_in[13];
    const float* hg_w    = (const float*)d_in[14];
    const float* wh_w    = (const float*)d_in[15];
    const float* wh_b    = (const float*)d_in[16];
    const float* norm_g  = (const float*)d_in[17];
    const float* norm_b  = (const float*)d_in[18];
    const float* tq_w    = (const float*)d_in[19];
    const float* tq_b    = (const float*)d_in[20];
    const float* tk_w    = (const float*)d_in[21];
    const float* tk_b    = (const float*)d_in[22];
    const float* tv_w    = (const float*)d_in[23];
    const float* tv_b    = (const float*)d_in[24];
    const float* ao_w    = (const float*)d_in[25];
    const float* ao_b    = (const float*)d_in[26];
    const float* n1_g    = (const float*)d_in[27];
    const float* n1_b    = (const float*)d_in[28];
    const float* f1_w    = (const float*)d_in[29];
    const float* f1_b    = (const float*)d_in[30];
    const float* f2_w    = (const float*)d_in[31];
    const float* f2_b    = (const float*)d_in[32];
    const float* n2_g    = (const float*)d_in[33];
    const float* n2_b    = (const float*)d_in[34];
    (void)in_sizes; (void)n_in; (void)out_size; (void)ws_size;

    // ---- workspace layout -------------------------------------------------
    char* W = (char*)d_ws;
    size_t cur = 0;
    auto allocF = [&](size_t n) {
        float* p = (float*)(W + cur);
        cur = (cur + n * sizeof(float) + 255) & ~(size_t)255;
        return p;
    };
    auto allocH = [&](size_t n) {
        _Float16* p = (_Float16*)(W + cur);
        cur = (cur + n * sizeof(_Float16) + 255) & ~(size_t)255;
        return p;
    };

    const size_t SZ_MD  = (size_t)M_ * D_;            // 3,145,728
    const size_t SZ_DD  = (size_t)D_ * D_;            // 589,824
    const size_t SZ_DF  = (size_t)D_ * FF_;           // 1,179,648
    const size_t SZ_SC  = (size_t)B_ * NH_ * L_ * L_; // 12,582,912
    const size_t SZ_HM  = (size_t)B_ * L_ * 2 * L_;   // 2,097,152
    const size_t SZ_HTX = (size_t)B_ * 2 * L_ * D_;   // 6,291,456
    const size_t SZ_FF  = (size_t)M_ * FF_;           // 6,291,456

    float* hA     = allocF(SZ_MD);
    float* hB     = allocF(SZ_MD);
    float* t0     = allocF(SZ_MD);
    float* t1     = allocF(SZ_MD);
    float* t2     = allocF(SZ_MD);
    float* kf     = allocF(SZ_MD);
    float* scores = allocF(SZ_SC);
    float* wamean = allocF((size_t)B_ * L_ * L_);
    float* Hm     = allocF(SZ_HM);
    float* HTX    = allocF(SZ_HTX);
    float* ff1    = allocF(SZ_FF);
    float* aspect = allocF((size_t)B_ * D_);
    float* asp    = allocF((size_t)B_ * DK_);
    float* aw     = allocF((size_t)B_ * NH_ * DK_);
    float* ascv   = allocF((size_t)B_ * NH_ * L_);
    float* Dv     = allocF((size_t)B_ * L_);
    float* De     = allocF((size_t)B_ * 2 * L_);

    // transposed f16 weights (Bt = N x K row-major)
    _Float16* winT = allocH(SZ_DD);
    _Float16* qwT  = allocH(SZ_DD);
    _Float16* kwT  = allocH(SZ_DD);
    _Float16* hgT  = allocH(NL_ * SZ_DD);
    _Float16* whT  = allocH(NL_ * SZ_DD);
    _Float16* tqT  = allocH(NL_ * SZ_DD);
    _Float16* tkT  = allocH(NL_ * SZ_DD);
    _Float16* tvT  = allocH(NL_ * SZ_DD);
    _Float16* aoT  = allocH(NL_ * SZ_DD);
    _Float16* f1T  = allocH(NL_ * SZ_DF);   // [NL][1536][768]
    _Float16* f2T  = allocH(NL_ * SZ_DF);   // [NL][768][1536]

    _Float16* h16    = allocH(SZ_MD);   // current h in f16 (A operand)
    _Float16* g16b   = allocH(SZ_MD);   // scratch f16
    _Float16* g16c   = allocH(SZ_MD);
    _Float16* g16d   = allocH(SZ_MD);
    _Float16* g16e   = allocH(SZ_MD);   // v2^T [B][D][L]
    _Float16* big16  = allocH(SZ_SC);   // softmaxed attention in f16 (A operand)
    _Float16* Hm16   = allocH(SZ_HM);   // A operand for HDe
    _Float16* HmT16  = allocH(SZ_HM);   // A operand for HTX
    _Float16* htxsT  = allocH(SZ_HTX);  // (De*HTX)^T [B][768][512]
    _Float16* ff16   = allocH(SZ_FF);   // relu(ff1) f16 (A operand)

    const long long LD = (long long)L_ * D_;   // 196608
    const long long LL = (long long)L_ * L_;   // 65536

    // ---- weight transpose-conversion fp32 -> f16 (Bt layout) ---------------
    cvt_t3(stream, winT, W_in, 1, D_, D_);
    cvt_t3(stream, qwT,  q_w,  1, D_, D_);
    cvt_t3(stream, kwT,  k_w,  1, D_, D_);
    cvt_t3(stream, hgT,  hg_w, NL_, D_, D_);
    cvt_t3(stream, whT,  wh_w, NL_, D_, D_);
    cvt_t3(stream, tqT,  tq_w, NL_, D_, D_);
    cvt_t3(stream, tkT,  tk_w, NL_, D_, D_);
    cvt_t3(stream, tvT,  tv_w, NL_, D_, D_);
    cvt_t3(stream, aoT,  ao_w, NL_, D_, D_);
    cvt_t3(stream, f1T,  f1_w, NL_, D_, FF_);   // -> [NL][1536][768]
    cvt_t3(stream, f2T,  f2_w, NL_, FF_, D_);   // -> [NL][768][1536]

    // ---- h = x @ W_in + b_in ----------------------------------------------
    cvt(stream, g16b, x, SZ_MD);
    run_gemm<true, false, 0>(stream, hA, g16b, winT, b_in, nullptr,
        M_, D_, D_, D_, D_, D_, 0,0, 0,0, 0,0, 0,0, 1, 1, 1.0f);
    cvt(stream, h16, hA, SZ_MD);

    // ---- aspect pipeline ---------------------------------------------------
    aspect_mean_kernel<<<(B_ * D_ + 255) / 256, 256, 0, stream>>>(aspect, hA, amask);

    run_gemm<true, false, 0>(stream, t0, h16, qwT, q_b, nullptr,
        M_, D_, D_, D_, D_, D_, 0,0, 0,0, 0,0, 0,0, 1, 1, 1.0f);
    cvt(stream, g16c, t0, SZ_MD);                       // q16 (A operand)
    run_gemm<true, false, 0>(stream, kf, h16, kwT, k_b, nullptr,
        M_, D_, D_, D_, D_, D_, 0,0, 0,0, 0,0, 0,0, 1, 1, 1.0f);
    cvt(stream, g16d, kf, SZ_MD);                       // k16 ([L,D] == Bt for q.k^T)

    asp_kernel<<<(B_ * DK_ + 255) / 256, 256, 0, stream>>>(asp, aspect, dense_w, dense_b);
    aw_kernel<<<(B_ * NH_ * DK_ + 255) / 256, 256, 0, stream>>>(aw, asp, weight_m);
    asc_kernel<<<(B_ * NH_ * L_ + 255) / 256, 256, 0, stream>>>(ascv, aw, kf, bias_m);

    // scores = q.k^T / 8  (batched 192; k16's [L,D] layout is Bt = N x K)
    run_gemm<false, false, 0>(stream, scores, g16c, g16d, nullptr, nullptr,
        L_, L_, DK_, D_, D_, L_,
        LD, 64, LD, 64, (long long)NH_ * LL, LL, 0, 0,
        NH_, B_ * NH_, 0.125f);
    softmax_kernel<<<B_ * NH_ * L_, L_, 0, stream>>>(scores, ascv, mask, 1);
    wamean_kernel<<<(int)(((long long)B_ * LL + 255) / 256), 256, 0, stream>>>(wamean, scores);

    // ---- layer loop --------------------------------------------------------
    float* hcur = hA;
    float* hnew = hB;
    for (int i = 0; i < NL_; ++i) {
        const _Float16* hgW = hgT + (size_t)i * SZ_DD;
        const _Float16* whW = whT + (size_t)i * SZ_DD;
        const _Float16* tqW = tqT + (size_t)i * SZ_DD;
        const _Float16* tkW = tkT + (size_t)i * SZ_DD;
        const _Float16* tvW = tvT + (size_t)i * SZ_DD;
        const _Float16* aoW = aoT + (size_t)i * SZ_DD;
        const _Float16* f1W = f1T + (size_t)i * SZ_DF;
        const _Float16* f2W = f2T + (size_t)i * SZ_DF;

        // hypergraph incidence + degree inverses
        build_hm_kernel<<<(int)((SZ_HM + 255) / 256), 256, 0, stream>>>(
            Hm, Hm16, HmT16, wamean, adj);
        dv_kernel<<<(B_ * L_ + 255) / 256, 256, 0, stream>>>(Dv, Hm);
        de_kernel<<<(B_ * 2 * L_ + 255) / 256, 256, 0, stream>>>(De, Hm);

        // xw = h @ hg_w[i]
        run_gemm<false, false, 0>(stream, t0, h16, hgW, nullptr, nullptr,
            M_, D_, D_, D_, D_, D_, 0,0, 0,0, 0,0, 0,0, 1, 1, 1.0f);
        cvt_t3(stream, g16b, t0, B_, L_, D_);           // xw^T [B][768][256]

        // HTX = Hm^T @ xw   (batched 16; Bt = xw^T, ldb = 256)
        run_gemm<false, false, 0>(stream, HTX, HmT16, g16b, nullptr, nullptr,
            2 * L_, D_, L_, L_, L_, D_,
            (long long)2 * L_ * L_, 0, LD, 0, (long long)2 * L_ * D_, 0, 0, 0,
            1, B_, 1.0f);
        // (De*HTX)^T [B][768][512]
        scale_cvt_t3_kernel<<<2048, 256, 0, stream>>>(htxsT, HTX, De, B_, 2 * L_, D_);

        // HDe = Hm @ (De*HTX)   (batched 16; Bt = htxsT, ldb = 512)
        run_gemm<false, false, 0>(stream, t0, Hm16, htxsT, nullptr, nullptr,
            L_, D_, 2 * L_, 2 * L_, 2 * L_, D_,
            (long long)L_ * 2 * L_, 0, (long long)2 * L_ * D_, 0, LD, 0, 0, 0,
            1, B_, 1.0f);
        // hcpre = relu(Dv_inv * HDe) -> f16 (A operand, row-major)
        scale_cvt_kernel<true><<<2048, 256, 0, stream>>>(
            g16b, t0, Dv, D_, (long long)SZ_MD);

        // hc = relu(ln(hcpre @ wh_w + wh_b))
        run_gemm<true, false, 0>(stream, t1, g16b, whW, wh_b + i * D_, nullptr,
            M_, D_, D_, D_, D_, D_, 0,0, 0,0, 0,0, 0,0, 1, 1, 1.0f);
        ln_kernel<true><<<M_, 256, 0, stream>>>(t0, t1, nullptr,
            norm_g + i * D_, norm_b + i * D_, nullptr);   // t0 = hc (f32)
        cvt(stream, g16b, t0, SZ_MD);                     // hc16 (A operand)

        // q2/k2/v2
        run_gemm<true, false, 0>(stream, t1, g16b, tqW, tq_b + i * D_, nullptr,
            M_, D_, D_, D_, D_, D_, 0,0, 0,0, 0,0, 0,0, 1, 1, 1.0f);
        cvt(stream, g16c, t1, SZ_MD);                     // q2 f16 (A)
        run_gemm<true, false, 0>(stream, t1, g16b, tkW, tk_b + i * D_, nullptr,
            M_, D_, D_, D_, D_, D_, 0,0, 0,0, 0,0, 0,0, 1, 1, 1.0f);
        cvt(stream, g16d, t1, SZ_MD);                     // k2 f16 (Bt for s2)
        run_gemm<true, false, 0>(stream, t1, g16b, tvW, tv_b + i * D_, nullptr,
            M_, D_, D_, D_, D_, D_, 0,0, 0,0, 0,0, 0,0, 1, 1, 1.0f);
        cvt_t3(stream, g16e, t1, B_, L_, D_);             // v2^T [B][768][256]

        // s2 = q2.k2^T / 8 -> softmax (mask only)
        run_gemm<false, false, 0>(stream, scores, g16c, g16d, nullptr, nullptr,
            L_, L_, DK_, D_, D_, L_,
            LD, 64, LD, 64, (long long)NH_ * LL, LL, 0, 0,
            NH_, B_ * NH_, 0.125f);
        softmax_kernel<<<B_ * NH_ * L_, L_, 0, stream>>>(scores, nullptr, mask, 0);
        cvt(stream, big16, scores, (long long)SZ_SC);

        // ao = A @ v2 -> [B,L,D] interleaved heads
        // Bt = v2^T slice per (b,h): offset b*D*L + (h*64)*L, ldb = 256
        run_gemm<false, false, 0>(stream, t1, big16, g16e, nullptr, nullptr,
            L_, DK_, L_, L_, L_, D_,
            (long long)NH_ * LL, LL, LD, (long long)64 * L_, LD, 64, 0, 0,
            NH_, B_ * NH_, 1.0f);
        cvt(stream, g16c, t1, SZ_MD);                     // ao f16 (A)

        // ao2 = ao @ ao_w + ao_b + hc -> ln(n1)
        run_gemm<true, true, 0>(stream, t2, g16c, aoW, ao_b + i * D_, t0,
            M_, D_, D_, D_, D_, D_, 0,0, 0,0, 0,0, 0,0, 1, 1, 1.0f);
        ln_kernel<false><<<M_, 256, 0, stream>>>(t1, t2, nullptr,
            n1_g + i * D_, n1_b + i * D_, nullptr);       // t1 = aoLN
        cvt(stream, g16c, t1, SZ_MD);                     // aoLN f16 (A)

        // ff = relu(aoLN @ f1 + b1) @ f2 + b2 + aoLN
        run_gemm<true, false, 1>(stream, ff1, g16c, f1W, f1_b + i * FF_, nullptr,
            M_, FF_, D_, D_, D_, FF_, 0,0, 0,0, 0,0, 0,0, 1, 1, 1.0f);
        cvt(stream, ff16, ff1, (long long)SZ_FF);
        run_gemm<true, true, 0>(stream, t2, ff16, f2W, f2_b + i * D_, t1,
            M_, D_, FF_, FF_, FF_, D_, 0,0, 0,0, 0,0, 0,0, 1, 1, 1.0f);

        // h = ln(ff + aoLN, n2) + h0
        ln_kernel<false><<<M_, 256, 0, stream>>>(hnew, t2, nullptr,
            n2_g + i * D_, n2_b + i * D_, hcur);
        cvt(stream, h16, hnew, SZ_MD);

        float* tmp = hcur; hcur = hnew; hnew = tmp;
    }

    // ---- output ------------------------------------------------------------
    copy_kernel<<<2048, 256, 0, stream>>>((float*)d_out, hcur, (long long)SZ_MD);
}
